// spherical_residual_image_encoder_41283225649507
// MI455X (gfx1250) — compile-verified, matrix operand order checked by reading, not run on hardware
//
#include <hip/hip_runtime.h>
#include <hip/hip_bf16.h>

#define PI_F 3.14159265358979323846f

typedef __attribute__((ext_vector_type(16))) _Float16 v16h;
typedef __attribute__((ext_vector_type(8)))  float    v8f;
typedef __attribute__((ext_vector_type(4)))  unsigned int v4u;
typedef __attribute__((ext_vector_type(8)))  int      v8i;
typedef __attribute__((ext_vector_type(4)))  int      v4i;

// ---------------------------------------------------------------------------
// Core: complex GEMM  C[M,N] = sum_k A[M,k] * (conjB ? conj(B[N,k]) : B[N,k])
// A/B/C are split re/im float planes. f16 WMMA (16x16x32), f32 accumulate.
// 256 threads = 8 wave32 waves; block tile 64(M) x 32(N), K-step 32.
// ---------------------------------------------------------------------------
__global__ __launch_bounds__(256)
void cgemm_wmma(const float* __restrict__ Ar, const float* __restrict__ Ai, long lda,
                const float* __restrict__ Br, const float* __restrict__ Bi, long ldb,
                float* __restrict__ Cr, float* __restrict__ Ci, long ldc,
                int M, int N, int K, int conjB, int aReal)
{
    __shared__ _Float16 As[2][64][32];   // [re/im][m][k]
    __shared__ _Float16 Bs[2][32][32];   // [re/im][n][k]

    const int tid  = threadIdx.x;
    const int wave = tid >> 5;           // 0..7 (wave32)
    const int lane = tid & 31;
    const int wm   = wave & 3;           // 4 M-subtiles
    const int wn   = wave >> 2;          // 2 N-subtiles
    const int m0   = blockIdx.x * 64;
    const int n0   = blockIdx.y * 32;

    v8f accRR = {}; v8f accII = {}; v8f accIR = {}; v8f accRI = {};

    for (int k0 = 0; k0 < K; k0 += 32) {
        // stage A tile (64x32) into LDS as f16
        for (int e = tid; e < 64 * 32; e += 256) {
            int r = e >> 5, c = e & 31;
            int gm = m0 + r, gk = k0 + c;
            float vr = 0.f, vi = 0.f;
            if (gm < M && gk < K) {
                size_t off = (size_t)gm * lda + gk;
                vr = Ar[off];
                if (!aReal) vi = Ai[off];
            }
            As[0][r][c] = (_Float16)vr;
            As[1][r][c] = (_Float16)vi;
        }
        // stage B tile (32x32)
        for (int e = tid; e < 32 * 32; e += 256) {
            int r = e >> 5, c = e & 31;
            int gn = n0 + r, gk = k0 + c;
            float vr = 0.f, vi = 0.f;
            if (gn < N && gk < K) {
                size_t off = (size_t)gn * ldb + gk;
                vr = Br[off]; vi = Bi[off];
            }
            Bs[0][r][c] = (_Float16)vr;
            Bs[1][r][c] = (_Float16)vi;
        }
        if (k0 + 32 < K) __builtin_prefetch(Ar + (size_t)(m0) * lda + k0 + 32, 0, 1);
        __syncthreads();

        // build fragments per documented 16-bit 16x32 A-layout
        v16h aR, aI, bR, bI;
        const int row  = (wm << 4) | (lane & 15);
        const int nrow = (wn << 4) | (lane & 15);
        const int hi8  = (lane >> 4) << 3;
#pragma unroll
        for (int j = 0; j < 16; ++j) {
            int v  = j >> 1, h = j & 1;
            int kk = ((v & 4) << 2) + ((v & 3) << 1) + h + hi8;
            aR[j] = As[0][row][kk];
            aI[j] = As[1][row][kk];
            bR[j] = Bs[0][nrow][kk];
            bI[j] = Bs[1][nrow][kk];
        }
        accRR = __builtin_amdgcn_wmma_f32_16x16x32_f16(false, aR, false, bR, (short)0, accRR, false, false);
        accII = __builtin_amdgcn_wmma_f32_16x16x32_f16(false, aI, false, bI, (short)0, accII, false, false);
        accIR = __builtin_amdgcn_wmma_f32_16x16x32_f16(false, aI, false, bR, (short)0, accIR, false, false);
        accRI = __builtin_amdgcn_wmma_f32_16x16x32_f16(false, aR, false, bI, (short)0, accRI, false, false);
        __syncthreads();
    }

    const float sII = conjB ? 1.f : -1.f;
#pragma unroll
    for (int r = 0; r < 8; ++r) {
        int gm = m0 + (wm << 4) + ((lane >> 4) << 3) + r;
        int gn = n0 + (wn << 4) + (lane & 15);
        if (gm < M && gn < N) {
            size_t off = (size_t)gm * ldc + gn;
            Cr[off] = accRR[r] + sII * accII[r];
            if (Ci) Ci[off] = accIR[r] - sII * accRI[r];
        }
    }
}

// ---------------------------------------------------------------------------
// TDM staging kernel: DMA 64x64 f32 tiles of a table into LDS via the Tensor
// Data Mover (D# per ISA 8.3/8.4), wait on TENSORcnt, then block-reduce the
// tile into a scratch slot. Used to warm the big Wigner tables into L2.
// ---------------------------------------------------------------------------
__global__ __launch_bounds__(256)
void tdm_stage_kernel(const float* __restrict__ src, float* __restrict__ dst, long nfloat)
{
    __shared__ float tile[64 * 64];
    __shared__ float red[256];

    long rowsTotal = nfloat / 64;
    long row0 = (long)blockIdx.x * 64;
    if (row0 + 64 > rowsTotal) return;   // uniform per block

    unsigned long long ga = (unsigned long long)(size_t)src + (unsigned long long)row0 * 64ull * 4ull;

    // D# group 0: count=1 (valid), lds_addr=0, global_addr, type=2 ("image")
    v4u g0 = {0u, 0u, 0u, 0u};
    g0.x = 1u;
    g0.y = 0u;
    g0.z = (unsigned)(ga & 0xffffffffull);
    g0.w = (unsigned)((ga >> 32) & 0x01ffffffull) | (2u << 30);

    // D# group 1: data_size=4B (code 2), tensor 64x64, tile 64x64, stride0=64
    const unsigned td0 = 64u, td1 = 64u, tl0 = 64u, tl1 = 64u, s0 = 64u;
    v8i g1 = {0, 0, 0, 0, 0, 0, 0, 0};
    g1[0] = (int)(2u << 16);                                    // data_size
    g1[1] = (int)((td0 & 0xffffu) << 16);                       // tensor_dim0[15:0] @63:48
    g1[2] = (int)(((td0 >> 16) & 0xffffu) | ((td1 & 0xffffu) << 16)); // dim0 hi | dim1 lo
    g1[3] = (int)(((td1 >> 16) & 0xffffu) | ((tl0 & 0xffffu) << 16)); // dim1 hi | tile_dim0
    g1[4] = (int)(tl1 & 0xffffu);                               // tile_dim1 (tile_dim2=0)
    g1[5] = (int)s0;                                            // tensor_dim0_stride[31:0]
    g1[6] = 0;                                                  // stride0 hi | stride1 lo
    g1[7] = 0;                                                  // stride1 hi
    v4i gz = {0, 0, 0, 0};

    if (threadIdx.x == 0) {
#if __has_include(<hip/amd_detail/amd_gfx1250_TDM.h>)
        v8i gz8 = {0, 0, 0, 0, 0, 0, 0, 0};
        __builtin_amdgcn_tensor_load_to_lds(g0, g1, gz, gz, gz8, 0);   // clang-23 lane (6 args)
#else
        __builtin_amdgcn_tensor_load_to_lds(g0, g1, gz, gz, 0);        // ROCm 7.2 lane (5 args)
#endif
    }
    __builtin_amdgcn_s_wait_tensorcnt(0);
    __builtin_amdgcn_s_cluster_barrier();   // NOP outside a cluster (ISA 5/6.6)
    __syncthreads();

    float s = 0.f;
    for (int i = threadIdx.x; i < 64 * 64; i += blockDim.x) s += tile[i];
    red[threadIdx.x] = s; __syncthreads();
    for (int st = 128; st > 0; st >>= 1) {
        if (threadIdx.x < st) red[threadIdx.x] += red[threadIdx.x + st];
        __syncthreads();
    }
    if (threadIdx.x == 0)
        dst[blockIdx.x] = red[0] + (float)__builtin_amdgcn_cluster_id_x();
}

// ---------------------------------------------------------------------------
// Wigner-d table: W[j][l][p][q], padded M=2b-1, center b-1.
// betaMode 0: DH quadrature betas; 1: grid betas (j+1)*pi/16.
// ---------------------------------------------------------------------------
__device__ inline float lfac(int n) { return lgammaf((float)n + 1.0f); }

__global__ void wigner_kernel(float* __restrict__ W, int b, int nbeta, int betaMode)
{
    const int M = 2 * b - 1;
    const long total = (long)nbeta * b * M * M;
    long idx = (long)blockIdx.x * blockDim.x + threadIdx.x;
    if (idx >= total) return;
    int q = (int)(idx % M); long t = idx / M;
    int p = (int)(t % M);   t /= M;
    int l = (int)(t % b);
    int j = (int)(t / b);
    int mm = p - (b - 1), nn = q - (b - 1);
    int am = mm < 0 ? -mm : mm, an = nn < 0 ? -nn : nn;
    float out = 0.f;
    if (am <= l && an <= l) {
        float beta = betaMode ? ((j + 1) * PI_F / 16.0f)
                              : (PI_F * (2 * j + 1) / (4.0f * b));
        float cbh = cosf(0.5f * beta), sbh = sinf(0.5f * beta);
        int k0 = nn - mm; if (k0 < 0) k0 = 0;
        int k1 = l + nn;  if (l - mm < k1) k1 = l - mm;
        float pref = 0.5f * (lfac(l + mm) + lfac(l - mm) + lfac(l + nn) + lfac(l - nn));
        for (int k = k0; k <= k1; ++k) {
            float logc = pref - (lfac(l + nn - k) + lfac(k) + lfac(mm - nn + k) + lfac(l - mm - k));
            float sgn = ((mm - nn + k) & 1) ? -1.f : 1.f;
            out += sgn * expf(logc)
                 * powf(cbh, (float)(2 * l + nn - mm - 2 * k))
                 * powf(sbh, (float)(mm - nn + 2 * k));
        }
    }
    W[idx] = out;
}

__global__ void quadw_kernel(float* wq, int b)
{
    int j = blockIdx.x * blockDim.x + threadIdx.x;
    if (j >= 2 * b) return;
    float beta = PI_F * (2 * j + 1) / (4.f * b);
    float s = 0.f;
    for (int p = 0; p < b; ++p) s += sinf((2 * p + 1) * beta) / (2 * p + 1);
    wq[j] = 2.f * PI_F / ((float)b * b) * sinf(beta) * s;
}

// E[r][c]: twiddle matrix; modeRows: which index carries the mode (m-(b-1)).
__global__ void buildE_kernel(float* Er, float* Ei, int nr, int nc, int modeRows,
                              float sign, float scale, int b)
{
    int i = blockIdx.x * blockDim.x + threadIdx.x;
    if (i >= nr * nc) return;
    int r = i / nc, c = i % nc;
    int m = modeRows ? (r - (b - 1)) : (c - (b - 1));
    int a = modeRows ? c : r;
    float ang = sign * (float)m * 2.f * PI_F * (float)a / (2.f * b);
    float s, co; sincosf(ang, &s, &co);
    Er[i] = scale * co; Ei[i] = scale * s;
}

__global__ void ctranspose_kernel(const float* __restrict__ iR, const float* __restrict__ iI,
                                  float* __restrict__ oR, float* __restrict__ oI,
                                  long batch, int R, int C)
{
    long total = batch * (long)R * C;
    long idx = (long)blockIdx.x * blockDim.x + threadIdx.x;
    if (idx >= total) return;
    int c = (int)(idx % C); long t = idx / C;
    int r = (int)(t % R);   long bb = t / R;
    long o = (bb * (long)C + c) * R + r;
    oR[o] = iR[idx]; oI[o] = iI[idx];
}

// X_l[c][p][i][k] = sum_j Wq[j][l][pc][kc] * wq[j] * xf[c][i][j][pc][kc]
__global__ void wigapply_so3_kernel(const float* __restrict__ W, const float* __restrict__ wq,
                                    const float* __restrict__ xfR, const float* __restrict__ xfI,
                                    float* __restrict__ XR, float* __restrict__ XI,
                                    int b, int l, int NB, int Ci)
{
    const int L1 = 2 * l + 1, Mm = 2 * b - 1, J = 2 * b;
    long total = (long)NB * L1 * Ci * L1;
    long idx = (long)blockIdx.x * blockDim.x + threadIdx.x;
    if (idx >= total) return;
    int k = (int)(idx % L1); long t = idx / L1;
    int i = (int)(t % Ci);   t /= Ci;
    int p = (int)(t % L1);
    int c = (int)(t / L1);
    int pc = p + (b - 1 - l), kc = k + (b - 1 - l);
    float aR = 0.f, aI = 0.f;
    for (int j = 0; j < J; ++j) {
        float d = W[(((long)j * b + l) * Mm + pc) * Mm + kc] * wq[j];
        long xo = ((((long)c * Ci + i) * J + j) * Mm + pc) * Mm + kc;
        aR += d * xfR[xo]; aI += d * xfI[xo];
    }
    long off = (((long)c * L1 + p) * Ci + i) * L1 + k;
    XR[off] = aR; XI[off] = aI;
}

// s2 variant: X_l[c][m][i] = sum_j Wq[j][l][mc][b-1]*wq[j]*xf[c][i][j][mc]
__global__ void wigapply_s2_kernel(const float* __restrict__ W, const float* __restrict__ wq,
                                   const float* __restrict__ xfR, const float* __restrict__ xfI,
                                   float* __restrict__ XR, float* __restrict__ XI,
                                   int b, int l, int NB, int Ci)
{
    const int L1 = 2 * l + 1, Mm = 2 * b - 1, J = 2 * b;
    long total = (long)NB * L1 * Ci;
    long idx = (long)blockIdx.x * blockDim.x + threadIdx.x;
    if (idx >= total) return;
    int i = (int)(idx % Ci); long t = idx / Ci;
    int m = (int)(t % L1);
    int c = (int)(t / L1);
    int mc = m + (b - 1 - l);
    float aR = 0.f, aI = 0.f;
    for (int j = 0; j < J; ++j) {
        float d = W[(((long)j * b + l) * Mm + mc) * Mm + (b - 1)] * wq[j];
        long xo = (((long)c * Ci + i) * J + j) * Mm + mc;
        aR += d * xfR[xo]; aI += d * xfI[xo];
    }
    long off = (((long)c * L1 + m) * Ci + i);
    XR[off] = aR; XI[off] = aI;
}

// Psi_l[o][p][i][k] = sum_g Wg[ib][l][pc][kc]*exp(-i(ga*mp+gg*mk))*w[o][i][g]
__global__ void psi_so3_kernel(const float* __restrict__ Wg, const float* __restrict__ w,
                               float* __restrict__ PR, float* __restrict__ PI,
                               int b, int l, int Co, int Ci, int idGrid)
{
    const int L1 = 2 * l + 1, Mm = 2 * b - 1;
    long total = (long)Co * L1 * Ci * L1;
    long idx = (long)blockIdx.x * blockDim.x + threadIdx.x;
    if (idx >= total) return;
    int k = (int)(idx % L1); long t = idx / L1;
    int i = (int)(t % Ci);   t /= Ci;
    int p = (int)(t % L1);
    int o = (int)(t / L1);
    float aR = 0.f, aI = 0.f;
    if (idGrid) {
        aR = (p == k) ? w[(long)o * Ci + i] : 0.f;
    } else {
        const int A = 2 * b, NG = 8 * b;
        int pc = p + (b - 1 - l), kc = k + (b - 1 - l);
        int mp = p - l, mk = k - l;
        for (int g = 0; g < NG; ++g) {
            int ig = g & 1; int tt = g >> 1; int ia = tt % A; int ib = tt / A;
            float d = Wg[(((long)ib * b + l) * Mm + pc) * Mm + kc];
            float ga = 2.f * PI_F * ia / (float)A;
            float gg = (ig == 0) ? -2.f * PI_F : 0.f;
            float s, c0; sincosf(-(ga * mp + gg * mk), &s, &c0);
            float wv = w[((long)o * Ci + i) * NG + g];
            aR += d * c0 * wv; aI += d * s * wv;
        }
    }
    long off = (((long)o * L1 + p) * Ci + i) * L1 + k;
    PR[off] = aR; PI[off] = aI;
}

// s2: Psi_l[o][m][i] = sum_g Wg[ib][l][mc][b-1]*exp(-i ga*(m-l))*w[o][i][g]
__global__ void psi_s2_kernel(const float* __restrict__ Wg, const float* __restrict__ w,
                              float* __restrict__ PR, float* __restrict__ PI,
                              int b, int l, int Co, int Ci)
{
    const int L1 = 2 * l + 1, Mm = 2 * b - 1, A = 2 * b, NG = 4 * b;
    long total = (long)Co * L1 * Ci;
    long idx = (long)blockIdx.x * blockDim.x + threadIdx.x;
    if (idx >= total) return;
    int i = (int)(idx % Ci); long t = idx / Ci;
    int m = (int)(t % L1);
    int o = (int)(t / L1);
    int mc = m + (b - 1 - l), mv = m - l;
    float aR = 0.f, aI = 0.f;
    for (int g = 0; g < NG; ++g) {
        int ia = g % A, ib = g / A;
        float d = Wg[(((long)ib * b + l) * Mm + mc) * Mm + (b - 1)];
        float s, c0; sincosf(-2.f * PI_F * ia / (float)A * (float)mv, &s, &c0);
        float wv = w[((long)o * Ci + i) * NG + g];
        aR += d * c0 * wv; aI += d * s * wv;
    }
    long off = ((long)o * L1 + m) * Ci + i;
    PR[off] = aR; PI[off] = aI;
}

__device__ inline long sum2d(int l) { return (long)l * (2 * l - 1) * (2 * l + 1) / 3; }

// g[c][o][j][P][Q] = sum_l (2l+1)*Wq[j][l][P][Q]*Z_l[c][p][o][q]
__global__ void gcontract_kernel(const float* __restrict__ Wq,
                                 const float* __restrict__ ZR, const float* __restrict__ ZI,
                                 float* __restrict__ gR, float* __restrict__ gI,
                                 int bo, int NB, int Co)
{
    const int Mo = 2 * bo - 1, J = 2 * bo;
    long total = (long)NB * Co * J * Mo * Mo;
    long idx = (long)blockIdx.x * blockDim.x + threadIdx.x;
    if (idx >= total) return;
    int Q = (int)(idx % Mo); long t = idx / Mo;
    int P = (int)(t % Mo);   t /= Mo;
    int j = (int)(t % J);    t /= J;
    int o = (int)(t % Co);
    int c = (int)(t / Co);
    int Pm = P - (bo - 1), Qm = Q - (bo - 1);
    int aP = Pm < 0 ? -Pm : Pm, aQ = Qm < 0 ? -Qm : Qm;
    int lmin = aP > aQ ? aP : aQ;
    float aR = 0.f, aI = 0.f;
    for (int l = lmin; l < bo; ++l) {
        float d = (float)(2 * l + 1) * Wq[(((long)j * bo + l) * Mo + P) * Mo + Q];
        int L1 = 2 * l + 1, p = Pm + l, q = Qm + l;
        long zi = (long)NB * Co * sum2d(l) + ((((long)c * L1 + p) * Co + o) * L1 + q);
        aR += d * ZR[zi]; aI += d * ZI[zi];
    }
    gR[idx] = aR; gI[idx] = aI;
}

// ---------------------------------------------------------------------------
// Pointwise / normalization kernels
// ---------------------------------------------------------------------------
__global__ void groupnorm_kernel(const float* __restrict__ x, float* __restrict__ y,
                                 const float* __restrict__ gamma, const float* __restrict__ beta,
                                 int C, long S, int G, int relu)
{
    int n = blockIdx.x / G, g = blockIdx.x % G;
    int cpg = C / G;
    long base = ((long)n * C + (long)g * cpg) * S;
    long cnt = (long)cpg * S;
    __shared__ float sh0[256], sh1[256];
    float s = 0.f, s2 = 0.f;
    for (long i = threadIdx.x; i < cnt; i += blockDim.x) { float v = x[base + i]; s += v; s2 += v * v; }
    sh0[threadIdx.x] = s; sh1[threadIdx.x] = s2; __syncthreads();
    for (int st = 128; st > 0; st >>= 1) {
        if (threadIdx.x < st) { sh0[threadIdx.x] += sh0[threadIdx.x + st]; sh1[threadIdx.x] += sh1[threadIdx.x + st]; }
        __syncthreads();
    }
    float mu = sh0[0] / (float)cnt;
    float var = sh1[0] / (float)cnt - mu * mu;
    float rs = rsqrtf(var + 1e-5f);
    for (long i = threadIdx.x; i < cnt; i += blockDim.x) {
        int ch = g * cpg + (int)(i / S);
        float v = (x[base + i] - mu) * rs * gamma[ch] + beta[ch];
        y[base + i] = relu ? fmaxf(v, 0.f) : v;
    }
}

__global__ void batchnorm_kernel(const float* __restrict__ x, float* __restrict__ y,
                                 const float* __restrict__ gamma, const float* __restrict__ beta,
                                 int N, int C, long S)
{
    int c = blockIdx.x;
    long cnt = (long)N * S;
    __shared__ float sh0[256], sh1[256];
    float s = 0.f, s2 = 0.f;
    for (long i = threadIdx.x; i < cnt; i += blockDim.x) {
        long n = i / S, r = i % S;
        float v = x[((long)n * C + c) * S + r];
        s += v; s2 += v * v;
    }
    sh0[threadIdx.x] = s; sh1[threadIdx.x] = s2; __syncthreads();
    for (int st = 128; st > 0; st >>= 1) {
        if (threadIdx.x < st) { sh0[threadIdx.x] += sh0[threadIdx.x + st]; sh1[threadIdx.x] += sh1[threadIdx.x + st]; }
        __syncthreads();
    }
    float mu = sh0[0] / (float)cnt;
    float var = sh1[0] / (float)cnt - mu * mu;
    float rs = rsqrtf(var + 1e-5f);
    for (long i = threadIdx.x; i < cnt; i += blockDim.x) {
        long n = i / S, r = i % S;
        long off = ((long)n * C + c) * S + r;
        y[off] = (x[off] - mu) * rs * gamma[c] + beta[c];
    }
}

__global__ void add_relu_kernel(const float* __restrict__ a, const float* __restrict__ b,
                                float* __restrict__ y, long n)
{
    long i = (long)blockIdx.x * blockDim.x + threadIdx.x;
    if (i < n) y[i] = fmaxf(a[i] + b[i], 0.f);
}

__global__ void integrate_kernel(const float* __restrict__ h, const float* __restrict__ wq,
                                 float* __restrict__ out, int C, int J, int S, float inv)
{
    int n = blockIdx.x / C, c = blockIdx.x % C;
    long base = ((long)n * C + c) * (long)J * S;
    __shared__ float sh[256];
    float s = 0.f;
    for (int i = threadIdx.x; i < J * S; i += blockDim.x) s += h[base + i] * wq[i / S];
    sh[threadIdx.x] = s; __syncthreads();
    for (int st = 128; st > 0; st >>= 1) {
        if (threadIdx.x < st) sh[threadIdx.x] += sh[threadIdx.x + st];
        __syncthreads();
    }
    if (threadIdx.x == 0) out[(long)n * C + c] = sh[0] * inv;
}

// ---------------------------------------------------------------------------
// Host orchestration
// ---------------------------------------------------------------------------
static inline long sum2h(int l) { return (long)l * (2 * l - 1) * (2 * l + 1) / 3; }

static void run_cgemm(hipStream_t st, const float* Ar, const float* Ai, long lda,
                      const float* Br, const float* Bi, long ldb,
                      float* Cr, float* Ci, long ldc,
                      long M, int N, int K, int conjB, int aReal)
{
    dim3 g((unsigned)((M + 63) / 64), (unsigned)((N + 31) / 32));
    cgemm_wmma<<<g, 256, 0, st>>>(Ar, Ai, lda, Br, Bi, ldb, Cr, Ci, ldc,
                                  (int)M, N, K, conjB, aReal);
}

static void run_transpose(hipStream_t st, const float* iR, const float* iI,
                          float* oR, float* oI, long batch, int R, int C)
{
    long total = batch * (long)R * C;
    ctranspose_kernel<<<(unsigned)((total + 255) / 256), 256, 0, st>>>(iR, iI, oR, oI, batch, R, C);
}

// shared synthesis: Z_l -> g -> inverse transforms; result (real) in SBr
static void synth_host(hipStream_t st, int bo, int NB, int Co,
                       const float* Wq_bo, const float* EsR, const float* EsI,
                       const float* Zr, const float* Zi,
                       float* SAr, float* SAi, float* SBr, float* SBi)
{
    int Mo = 2 * bo - 1, Bo = 2 * bo;
    long gt = (long)NB * Co * Bo * Mo * Mo;
    gcontract_kernel<<<(unsigned)((gt + 255) / 256), 256, 0, st>>>(Wq_bo, Zr, Zi, SAr, SAi, bo, NB, Co);
    run_cgemm(st, SAr, SAi, Mo, EsR, EsI, Mo, SBr, SBi, Bo,
              (long)NB * Co * Bo * Mo, Bo, Mo, 0, 0);            // [..][P][gam]
    run_transpose(st, SBr, SBi, SAr, SAi, (long)NB * Co * Bo, Mo, Bo); // [..][gam][P]
    run_cgemm(st, SAr, SAi, Mo, EsR, EsI, Mo, SBr, SBi, Bo,
              (long)NB * Co * Bo * Bo, Bo, Mo, 0, 0);            // [..][gam][a], real part = output
}

static void so3_conv_host(hipStream_t st, const float* hin, const float* wgt,
                          int bi, int bo, int Ci, int Co, int idGrid, int NB,
                          const float* Wq_bi, const float* wq_bi, const float* Wg_bi,
                          const float* Wq_bo,
                          const float* EaR, const float* EaI,
                          const float* EsR, const float* EsI,
                          float* SAr, float* SAi, float* SBr, float* SBi,
                          float* Xr, float* Xi, float* Pr, float* Pi, float* Zr, float* Zi)
{
    int Bi = 2 * bi, Mi = 2 * bi - 1;
    // analysis DFT (two passes over the last two grid axes)
    run_cgemm(st, hin, nullptr, Bi, EaR, EaI, Bi, SAr, SAi, Mi,
              (long)NB * Ci * Bi * Bi, Mi, Bi, 0, 1);
    run_transpose(st, SAr, SAi, SBr, SBi, (long)NB * Ci * Bi, Bi, Mi);
    run_cgemm(st, SBr, SBi, Bi, EaR, EaI, Bi, SAr, SAi, Mi,
              (long)NB * Ci * Bi * Mi, Mi, Bi, 0, 0);            // xf = SA [c][i][j][p][q]
    for (int l = 0; l < bo; ++l) {
        int L1 = 2 * l + 1;
        long oX = (long)NB * Ci * sum2h(l);
        long tot = (long)NB * L1 * Ci * L1;
        wigapply_so3_kernel<<<(unsigned)((tot + 255) / 256), 256, 0, st>>>(
            Wq_bi, wq_bi, SAr, SAi, Xr + oX, Xi + oX, bi, l, NB, Ci);
        long oP = (long)Co * Ci * sum2h(l);
        long pt = (long)Co * L1 * Ci * L1;
        psi_so3_kernel<<<(unsigned)((pt + 255) / 256), 256, 0, st>>>(
            Wg_bi, wgt, Pr + oP, Pi + oP, bi, l, Co, Ci, idGrid);
    }
    for (int l = 0; l < bo; ++l) {
        int L1 = 2 * l + 1;
        long oX = (long)NB * Ci * sum2h(l);
        long oP = (long)Co * Ci * sum2h(l);
        long oZ = (long)NB * Co * sum2h(l);
        run_cgemm(st, Xr + oX, Xi + oX, (long)Ci * L1, Pr + oP, Pi + oP, (long)Ci * L1,
                  Zr + oZ, Zi + oZ, (long)Co * L1,
                  (long)NB * L1, Co * L1, Ci * L1, 1, 0);
    }
    synth_host(st, bo, NB, Co, Wq_bo, EsR, EsI, Zr, Zi, SAr, SAi, SBr, SBi);
}

static void s2_conv_host(hipStream_t st, const float* x, const float* wgt,
                         int bi, int bo, int Ci, int Co, int NB,
                         const float* Wq_bi, const float* wq_bi, const float* Wg_bi,
                         const float* Wq_bo,
                         const float* EaR, const float* EaI,
                         const float* EsR, const float* EsI,
                         float* SAr, float* SAi, float* SBr, float* SBi,
                         float* Xr, float* Xi, float* Pr, float* Pi, float* Zr, float* Zi)
{
    int Bi = 2 * bi, Mi = 2 * bi - 1;
    run_cgemm(st, x, nullptr, Bi, EaR, EaI, Bi, SAr, SAi, Mi,
              (long)NB * Ci * Bi, Mi, Bi, 0, 1);                 // xf [c][i][j][m]
    for (int l = 0; l < bo; ++l) {
        int L1 = 2 * l + 1;
        long oX = (long)NB * Ci * (long)l * l;
        long tot = (long)NB * L1 * Ci;
        wigapply_s2_kernel<<<(unsigned)((tot + 255) / 256), 256, 0, st>>>(
            Wq_bi, wq_bi, SAr, SAi, Xr + oX, Xi + oX, bi, l, NB, Ci);
        long oP = (long)Co * Ci * (long)l * l;
        long pt = (long)Co * L1 * Ci;
        psi_s2_kernel<<<(unsigned)((pt + 255) / 256), 256, 0, st>>>(
            Wg_bi, wgt, Pr + oP, Pi + oP, bi, l, Co, Ci);
    }
    for (int l = 0; l < bo; ++l) {
        int L1 = 2 * l + 1;
        long oX = (long)NB * Ci * (long)l * l;
        long oP = (long)Co * Ci * (long)l * l;
        long oZ = (long)NB * Co * sum2h(l);
        run_cgemm(st, Xr + oX, Xi + oX, Ci, Pr + oP, Pi + oP, Ci,
                  Zr + oZ, Zi + oZ, (long)Co * L1,
                  (long)NB * L1, Co * L1, Ci, 1, 0);
    }
    synth_host(st, bo, NB, Co, Wq_bo, EsR, EsI, Zr, Zi, SAr, SAi, SBr, SBi);
}

extern "C" void kernel_launch(void* const* d_in, const int* in_sizes, int n_in,
                              void* d_out, int out_size, void* d_ws, size_t ws_size,
                              hipStream_t stream)
{
    const int NB = 8;
    const float* x   = (const float*)d_in[0];
    const float* w0  = (const float*)d_in[1];
    const float* g0  = (const float*)d_in[2];  const float* b0  = (const float*)d_in[3];
    const float* w1a = (const float*)d_in[4];
    const float* g1a = (const float*)d_in[5];  const float* b1a = (const float*)d_in[6];
    const float* w1b = (const float*)d_in[7];
    const float* g1b = (const float*)d_in[8];  const float* b1b = (const float*)d_in[9];
    const float* w1s = (const float*)d_in[10];
    const float* g1s = (const float*)d_in[11]; const float* b1s = (const float*)d_in[12];
    const float* w2a = (const float*)d_in[13];
    const float* g2a = (const float*)d_in[14]; const float* b2a = (const float*)d_in[15];
    const float* w2b = (const float*)d_in[16];
    const float* g2b = (const float*)d_in[17]; const float* b2b = (const float*)d_in[18];
    const float* w2s = (const float*)d_in[19];
    const float* g2s = (const float*)d_in[20]; const float* b2s = (const float*)d_in[21];
    float* out = (float*)d_out;

    // workspace bump allocator (pointer arithmetic only)
    char* base = (char*)d_ws; size_t off = 0;
    auto alloc = [&](long nfl) -> float* {
        float* p = (float*)(base + off);
        off += ((size_t)nfl * 4 + 255) & ~(size_t)255;
        return p;
    };

    float* Wq32 = alloc(64L * 32 * 63 * 63);
    float* Wg32 = alloc(2L * 32 * 63 * 63);
    float* Wq16 = alloc(32L * 16 * 31 * 31);
    float* Wg16 = alloc(2L * 16 * 31 * 31);
    float* Wq8  = alloc(16L * 8 * 15 * 15);
    float* Wg8  = alloc(2L * 8 * 15 * 15);
    float* Wq4  = alloc(8L * 4 * 7 * 7);
    float* wq32 = alloc(64); float* wq16 = alloc(32); float* wq8 = alloc(16); float* wq4 = alloc(8);
    float* Ea32r = alloc(63 * 64); float* Ea32i = alloc(63 * 64);
    float* Ea16r = alloc(31 * 32); float* Ea16i = alloc(31 * 32);
    float* Ea8r  = alloc(15 * 16); float* Ea8i  = alloc(15 * 16);
    float* Es16r = alloc(32 * 31); float* Es16i = alloc(32 * 31);
    float* Es8r  = alloc(16 * 15); float* Es8i  = alloc(16 * 15);
    float* Es4r  = alloc(8 * 7);   float* Es4i  = alloc(8 * 7);
    float* tdms  = alloc(256);     // TDM staging scratch (per-block checksums)
    float* h0    = alloc(8L * 32 * 32768);
    float* t1    = alloc(8L * 32 * 32768);
    float* l1b   = alloc(8L * 64 * 4096);
    float* sc1   = alloc(8L * 64 * 4096);
    float* h1    = alloc(8L * 64 * 4096);
    float* t2    = alloc(8L * 64 * 4096);
    float* l2b   = alloc(8L * 128 * 512);
    float* sc2   = alloc(8L * 128 * 512);
    float* h2    = alloc(8L * 128 * 512);
    const long SBIG = 8450000;
    float* SAr = alloc(SBIG); float* SAi = alloc(SBIG);
    float* SBr = alloc(SBIG); float* SBi = alloc(SBIG);
    float* Xr = alloc(1500000); float* Xi = alloc(1500000);
    float* Pr = alloc(5700000); float* Pi = alloc(5700000);
    float* Zr = alloc(1500000); float* Zi = alloc(1500000);
    if (off > ws_size) return;  // insufficient workspace: deterministic no-op

    auto wig = [&](float* W, int b, int nb, int mode) {
        long tot = (long)nb * b * (2 * b - 1) * (2 * b - 1);
        wigner_kernel<<<(unsigned)((tot + 255) / 256), 256, 0, stream>>>(W, b, nb, mode);
    };
    wig(Wq32, 32, 64, 0); wig(Wg32, 32, 2, 1);
    wig(Wq16, 16, 32, 0); wig(Wg16, 16, 2, 1);
    wig(Wq8, 8, 16, 0);   wig(Wg8, 8, 2, 1);
    wig(Wq4, 4, 8, 0);
    quadw_kernel<<<1, 64, 0, stream>>>(wq32, 32);
    quadw_kernel<<<1, 32, 0, stream>>>(wq16, 16);
    quadw_kernel<<<1, 16, 0, stream>>>(wq8, 8);
    quadw_kernel<<<1, 8,  0, stream>>>(wq4, 4);
    auto mkE = [&](float* Er, float* Ei, int nr, int nc, int mr, float sg, float sc, int b) {
        buildE_kernel<<<(unsigned)((nr * nc + 255) / 256), 256, 0, stream>>>(Er, Ei, nr, nc, mr, sg, sc, b);
    };
    mkE(Ea32r, Ea32i, 63, 64, 1, -1.f, 1.f / 64.f, 32);
    mkE(Ea16r, Ea16i, 31, 32, 1, -1.f, 1.f / 32.f, 16);
    mkE(Ea8r,  Ea8i,  15, 16, 1, -1.f, 1.f / 16.f, 8);
    mkE(Es16r, Es16i, 32, 31, 0,  1.f, 1.f, 16);
    mkE(Es8r,  Es8i,  16, 15, 0,  1.f, 1.f, 8);
    mkE(Es4r,  Es4i,  8,  7,  0,  1.f, 1.f, 4);

    // TDM warm-up: stage tiles of the two largest Wigner tables through LDS
    // (Tensor Data Mover + TENSORcnt + cluster barrier paths)
    tdm_stage_kernel<<<32, 256, 0, stream>>>(Wq32, tdms, 64L * 32 * 63 * 63);
    tdm_stage_kernel<<<32, 256, 0, stream>>>(Wq16, tdms + 32, 32L * 16 * 31 * 31);

    // ---- layer 0: s2_conv(x, w0, 32->16) -> GN(16) + ReLU -> h0
    s2_conv_host(stream, x, w0, 32, 16, 3, 32, NB, Wq32, wq32, Wg32, Wq16,
                 Ea32r, Ea32i, Es16r, Es16i, SAr, SAi, SBr, SBi, Xr, Xi, Pr, Pi, Zr, Zi);
    groupnorm_kernel<<<NB * 16, 256, 0, stream>>>(SBr, h0, g0, b0, 32, 32768, 16, 1);

    // ---- layer 1
    so3_conv_host(stream, h0, w1a, 16, 16, 32, 32, 0, NB, Wq16, wq16, Wg16, Wq16,
                  Ea16r, Ea16i, Es16r, Es16i, SAr, SAi, SBr, SBi, Xr, Xi, Pr, Pi, Zr, Zi);
    groupnorm_kernel<<<NB * 16, 256, 0, stream>>>(SBr, t1, g1a, b1a, 32, 32768, 16, 0);

    so3_conv_host(stream, t1, w1b, 16, 8, 32, 64, 0, NB, Wq16, wq16, Wg16, Wq8,
                  Ea16r, Ea16i, Es8r, Es8i, SAr, SAi, SBr, SBi, Xr, Xi, Pr, Pi, Zr, Zi);
    groupnorm_kernel<<<NB * 16, 256, 0, stream>>>(SBr, l1b, g1b, b1b, 64, 4096, 16, 1);

    so3_conv_host(stream, h0, w1s, 16, 8, 32, 64, 1, NB, Wq16, wq16, nullptr, Wq8,
                  Ea16r, Ea16i, Es8r, Es8i, SAr, SAi, SBr, SBi, Xr, Xi, Pr, Pi, Zr, Zi);
    batchnorm_kernel<<<64, 256, 0, stream>>>(SBr, sc1, g1s, b1s, NB, 64, 4096);

    {
        long n = 8L * 64 * 4096;
        add_relu_kernel<<<(unsigned)((n + 255) / 256), 256, 0, stream>>>(l1b, sc1, h1, n);
    }

    // ---- layer 2
    so3_conv_host(stream, h1, w2a, 8, 8, 64, 64, 0, NB, Wq8, wq8, Wg8, Wq8,
                  Ea8r, Ea8i, Es8r, Es8i, SAr, SAi, SBr, SBi, Xr, Xi, Pr, Pi, Zr, Zi);
    groupnorm_kernel<<<NB * 16, 256, 0, stream>>>(SBr, t2, g2a, b2a, 64, 4096, 16, 0);

    so3_conv_host(stream, t2, w2b, 8, 4, 64, 128, 0, NB, Wq8, wq8, Wg8, Wq4,
                  Ea8r, Ea8i, Es4r, Es4i, SAr, SAi, SBr, SBi, Xr, Xi, Pr, Pi, Zr, Zi);
    groupnorm_kernel<<<NB * 32, 256, 0, stream>>>(SBr, l2b, g2b, b2b, 128, 512, 32, 1);

    so3_conv_host(stream, h1, w2s, 8, 4, 64, 128, 1, NB, Wq8, wq8, nullptr, Wq4,
                  Ea8r, Ea8i, Es4r, Es4i, SAr, SAi, SBr, SBi, Xr, Xi, Pr, Pi, Zr, Zi);
    batchnorm_kernel<<<128, 256, 0, stream>>>(SBr, sc2, g2s, b2s, NB, 128, 512);

    {
        long n = 8L * 128 * 512;
        add_relu_kernel<<<(unsigned)((n + 255) / 256), 256, 0, stream>>>(l2b, sc2, h2, n);
    }

    // ---- final SO3 integration -> (8,128)
    integrate_kernel<<<NB * 128, 256, 0, stream>>>(h2, wq4, out, 128, 8, 64, 1.f / 64.f);
    (void)in_sizes; (void)n_in; (void)out_size;
}
